// FeedbackGRUModel_46557445488727
// MI455X (gfx1250) — matrix operand order repaired and checked by reading, not compiled, not run on hardware
//
#include <hip/hip_runtime.h>

typedef __attribute__((ext_vector_type(16))) _Float16 v16h;
typedef __attribute__((ext_vector_type(8)))  float    v8f;

#define WAVES   16
#define THREADS 512

// Fragment storage offsets (in _Float16 elements). One block = 32 lanes * 16 halfs = 1KB.
#define FR_ENC 0        //  4 blocks (K=11->32 x1, N=64 ->4 tiles)   bias row at k=11
#define FR_WIH 2048     // 36 blocks (K=68->96 x3, N=192->12 tiles)  bias row at k=68
#define FR_WHH 20480    // 24 blocks (K=64 x2,     N=192->12 tiles)
#define FR_D1  32768    //  8 blocks (K=64 x2,     N=64 -> 4 tiles)
#define FR_D2  36864    //  2 blocks (K=64 x2,     N=4  -> 1 tile)
#define FR_TOTAL_HALF 37888
#define ACT_OFF_BYTES (FR_TOTAL_HALF * 2)          // 75776
#define ACT_PER_WAVE_F 3584                        // eny 16*96 + h0 16*64 + h1 16*64 floats
#define SHMEM_BYTES (ACT_OFF_BYTES + WAVES * ACT_PER_WAVE_F * 4)  // 305152 (<= 320KB WGP limit)

__device__ __forceinline__ v8f wmma16(v16h a, v16h b, v8f c) {
  return __builtin_amdgcn_wmma_f32_16x16x32_f16(false, a, false, b, (short)0, c, false, false);
}

__device__ __forceinline__ v8f bcast8(float v) {
  v8f c;
#pragma unroll
  for (int i = 0; i < 8; ++i) c[i] = v;
  return c;
}

__device__ __forceinline__ float relu1(float x) {
  return __builtin_amdgcn_fmed3f(x, 0.0f, __builtin_inff());  // single v_med3_f32
}
__device__ __forceinline__ float fast_rcp(float x) { return __builtin_amdgcn_rcpf(x); }
__device__ __forceinline__ float sigmoidf_fast(float x) { return fast_rcp(1.0f + __expf(-x)); }
__device__ __forceinline__ float tanhf_fast(float x) {
  float e = __expf(2.0f * x);
  return 1.0f - 2.0f * fast_rcp(e + 1.0f);
}

// Convert f32 weight matrix W[nOut][kIn] (row-major) into B-fragment layout:
// B[k][n] = W[n][k]; lane = column (n = nt*16 + lane%16),
// lanes 0-15 hold K={0..7,16..23}, lanes 16-31 hold K={8..15,24..31}, packed as f16.
__device__ __forceinline__ void fill_frags(_Float16* dst, const float* __restrict__ W,
                                           int nOut, int kIn, int numN, int numK, int tid) {
  int total = numN * numK * 512;
  for (int i = tid; i < total; i += THREADS) {
    int blk  = i >> 9;
    int rlo  = i & 511;
    int lane = rlo >> 4;
    int j    = rlo & 15;
    int ks   = blk / numN;
    int nt   = blk - ks * numN;
    int n    = nt * 16 + (lane & 15);
    int k    = ks * 32 + ((lane & 16) ? 8 : 0) + j + ((j & 8) ? 8 : 0);
    float v  = 0.0f;
    if (n < nOut && k < kIn) v = W[n * kIn + k];
    dst[(blk << 9) + (lane << 4) + j] = (_Float16)v;
  }
}

__device__ __forceinline__ v16h load_b_frag(const _Float16* frags, int blk, int lane) {
  return *(const v16h*)(frags + (blk << 9) + (lane << 4));
}

// A-fragment (16x32 f16) from per-wave f32 LDS activation buffer.
// lanes 0-15: row=lane,   K = koff+{0..7, 16..23}
// lanes16-31: row=lane-16,K = koff+{8..15,24..31}
__device__ __forceinline__ v16h load_a_frag(const float* base, int stride, int koff, int lane) {
  int r  = lane & 15;
  int kb = koff + ((lane & 16) ? 8 : 0);
  const float* p = base + r * stride + kb;
  float4 a0 = *(const float4*)(p);
  float4 a1 = *(const float4*)(p + 4);
  float4 a2 = *(const float4*)(p + 16);
  float4 a3 = *(const float4*)(p + 20);
  v16h a;
  a[0]=(_Float16)a0.x; a[1]=(_Float16)a0.y; a[2]=(_Float16)a0.z; a[3]=(_Float16)a0.w;
  a[4]=(_Float16)a1.x; a[5]=(_Float16)a1.y; a[6]=(_Float16)a1.z; a[7]=(_Float16)a1.w;
  a[8]=(_Float16)a2.x; a[9]=(_Float16)a2.y; a[10]=(_Float16)a2.z; a[11]=(_Float16)a2.w;
  a[12]=(_Float16)a3.x; a[13]=(_Float16)a3.y; a[14]=(_Float16)a3.z; a[15]=(_Float16)a3.w;
  return a;
}

// D fragment (16x16 f32): lane holds col n=lane%16, rows v + 8*(lane/16).
__device__ __forceinline__ void store_d(float* base, int stride, int n0, int lane, v8f d) {
  int n  = n0 + (lane & 15);
  int m0 = (lane & 16) ? 8 : 0;
#pragma unroll
  for (int v = 0; v < 8; ++v) base[(m0 + v) * stride + n] = d[v];
}

__device__ __forceinline__ v8f load_d(const float* base, int stride, int n0, int lane) {
  int n  = n0 + (lane & 15);
  int m0 = (lane & 16) ? 8 : 0;
  v8f d;
#pragma unroll
  for (int v = 0; v < 8; ++v) d[v] = base[(m0 + v) * stride + n];
  return d;
}

__global__ void __launch_bounds__(THREADS)
feedback_gru_kernel(const float* __restrict__ x,
                    const float* __restrict__ enc_w, const float* __restrict__ enc_b,
                    const float* __restrict__ ln_g,  const float* __restrict__ ln_b,
                    const float* __restrict__ w_ih,  const float* __restrict__ w_hh,
                    const float* __restrict__ b_ih,  const float* __restrict__ b_hh,
                    const float* __restrict__ dec1_w, const float* __restrict__ dec1_b,
                    const float* __restrict__ dec2_w, const float* __restrict__ dec2_b,
                    const float* __restrict__ y0,
                    float* __restrict__ out, int nrows) {
  extern __shared__ char smem[];
  _Float16* frags = (_Float16*)smem;
  float*    acts  = (float*)(smem + ACT_OFF_BYTES);

  const int tid  = threadIdx.x;
  const int lane = tid & 31;
  const int wave = tid >> 5;
  const int nl   = lane & 15;

  // One-time weight conversion into B-fragment layout (shared by all waves).
  fill_frags(frags + FR_ENC, enc_w,  64, 11,  4, 1, tid);
  fill_frags(frags + FR_WIH, w_ih,  192, 68, 12, 3, tid);
  fill_frags(frags + FR_WHH, w_hh,  192, 64, 12, 2, tid);
  fill_frags(frags + FR_D1,  dec1_w, 64, 64,  4, 2, tid);
  fill_frags(frags + FR_D2,  dec2_w,  4, 64,  1, 2, tid);
  __syncthreads();

  // Inject biases into the dead K rows so WMMA accumulators can start from inline 0:
  //  - WIH k=68 (ks=2, k%32=4 -> lanes 0-15, j=4): r/z tiles get b_ih+b_hh, n tiles get b_ih.
  //  - ENC k=11 (lanes 16-31, j=3): enc_b.  The matching A-side slots are set to 1.0.
  for (int n = tid; n < 192; n += THREADS) {
    float bv = (n < 128) ? (b_ih[n] + b_hh[n]) : b_ih[n];
    frags[FR_WIH + (24 + (n >> 4)) * 512 + ((n & 15) << 4) + 4] = (_Float16)bv;
  }
  for (int n = tid; n < 64; n += THREADS) {
    frags[FR_ENC + (n >> 4) * 512 + ((16 + (n & 15)) << 4) + 3] = (_Float16)enc_b[n];
  }
  __syncthreads();

  float* eny = acts + wave * ACT_PER_WAVE_F;  // [16][96] : enc 0..63, y 64..67, 1.0 at 68, pad
  float* h0  = eny + 16 * 96;                 // [16][64]
  float* h1  = h0  + 16 * 64;                 // [16][64]

  const int row0 = (blockIdx.x * WAVES + wave) * 16;
  __builtin_prefetch(x + (size_t)row0 * 11, 0, 0);

  // ---- Per-lane bias registers for tiles that cannot use the bias-in-K trick ----
  float bnh[4], bd1[4];
#pragma unroll
  for (int t = 0; t < 4; ++t) {
    bnh[t] = b_hh[(8 + t) * 16 + nl];
    bd1[t] = dec1_b[t * 16 + nl];
  }
  float bd2 = (nl < 4) ? dec2_b[nl] : 0.0f;

  // ---- Encoder: A-fragment straight from global x (K=11 bias slot, pad to 32) ----
  v16h ax;
  {
    int row = min(row0 + nl, nrows - 1);
    const float* xp = x + (size_t)row * 11;
#pragma unroll
    for (int j = 0; j < 16; ++j) ax[j] = (_Float16)0.0f;
    if (lane < 16) {
#pragma unroll
      for (int j = 0; j < 8; ++j) ax[j] = (_Float16)xp[j];      // K=0..7
    } else {
#pragma unroll
      for (int j = 0; j < 3; ++j) ax[j] = (_Float16)xp[8 + j];  // K=8..10
      ax[3] = (_Float16)1.0f;                                   // K=11 -> ones column
    }
  }
#pragma unroll
  for (int t = 0; t < 4; ++t) {
    v8f acc = {};
    acc = wmma16(ax, load_b_frag(frags + FR_ENC, t, lane), acc);
#pragma unroll
    for (int v = 0; v < 8; ++v) acc[v] = relu1(acc[v]);
    store_d(eny, 96, t * 16, lane, acc);
  }

  // ---- LayerNorm over 64 cols; init h=enc, y=y0, ones col 68, zero pad ----
  {
    int half = lane >> 4;
    const float* er = eny + nl * 96;
    float s = 0.0f, sq = 0.0f;
#pragma unroll
    for (int i = 0; i < 32; ++i) { float v = er[half * 32 + i]; s += v; sq += v * v; }
    s  += __shfl_xor(s, 16);
    sq += __shfl_xor(sq, 16);
    float mu  = s * (1.0f / 64.0f);
    float var = sq * (1.0f / 64.0f) - mu * mu;
    float rs  = rsqrtf(var + 1e-5f);
#pragma unroll
    for (int i = 0; i < 32; ++i) {
      int c = half * 32 + i;
      float v = (er[c] - mu) * rs * ln_g[c] + ln_b[c];
      eny[nl * 96 + c] = v;
      h0[nl * 64 + c]  = v;
    }
#pragma unroll
    for (int i = 0; i < 16; ++i) {             // cols 64..95: y0, ones col, zero pad
      int c = 64 + half * 16 + i;
      float v = 0.0f;
      if (c < 68) v = y0[c - 64];
      else if (c == 68) v = 1.0f;
      eny[nl * 96 + c] = v;
    }
  }

  // enc part of eny (K-steps 0,1) never changes after LN -> hoist out of the loop.
  v16h ae0 = load_a_frag(eny, 96, 0,  lane);
  v16h ae1 = load_a_frag(eny, 96, 32, lane);

  float* hcur  = h0;
  float* hnext = h1;

  for (int it = 0; it < 5; ++it) {
    // A-fragments for this iteration (y changed -> reload K-step 2; h changed -> reload both)
    v16h ae2 = load_a_frag(eny, 96, 64, lane);
    v16h ah0 = load_a_frag(hcur, 64, 0,  lane);
    v16h ah1 = load_a_frag(hcur, 64, 32, lane);

    // ---- r gate (tiles 0..3): gi + gh summed; bias comes in via K=68 ----
    v8f r[4];
#pragma unroll
    for (int t = 0; t < 4; ++t) {
      v8f acc = {};
      acc = wmma16(ae0, load_b_frag(frags + FR_WIH, t,      lane), acc);
      acc = wmma16(ae1, load_b_frag(frags + FR_WIH, 12 + t, lane), acc);
      acc = wmma16(ae2, load_b_frag(frags + FR_WIH, 24 + t, lane), acc);
      acc = wmma16(ah0, load_b_frag(frags + FR_WHH, t,      lane), acc);
      acc = wmma16(ah1, load_b_frag(frags + FR_WHH, 12 + t, lane), acc);
#pragma unroll
      for (int v = 0; v < 8; ++v) acc[v] = sigmoidf_fast(acc[v]);
      r[t] = acc;
    }

    // ---- z and n gates per tile; h update ----
#pragma unroll
    for (int t = 0; t < 4; ++t) {
      // z tile (4+t)
      v8f zt = {};
      zt = wmma16(ae0, load_b_frag(frags + FR_WIH, 4 + t,  lane), zt);
      zt = wmma16(ae1, load_b_frag(frags + FR_WIH, 16 + t, lane), zt);
      zt = wmma16(ae2, load_b_frag(frags + FR_WIH, 28 + t, lane), zt);
      zt = wmma16(ah0, load_b_frag(frags + FR_WHH, 4 + t,  lane), zt);
      zt = wmma16(ah1, load_b_frag(frags + FR_WHH, 16 + t, lane), zt);
#pragma unroll
      for (int v = 0; v < 8; ++v) zt[v] = sigmoidf_fast(zt[v]);

      // inn tile (8+t), bias via K=68
      v8f ai = {};
      ai = wmma16(ae0, load_b_frag(frags + FR_WIH, 8 + t,  lane), ai);
      ai = wmma16(ae1, load_b_frag(frags + FR_WIH, 20 + t, lane), ai);
      ai = wmma16(ae2, load_b_frag(frags + FR_WIH, 32 + t, lane), ai);
      // hn tile (8+t), separate accumulator with b_hh bias
      v8f ag = bcast8(bnh[t]);
      ag = wmma16(ah0, load_b_frag(frags + FR_WHH, 8 + t,  lane), ag);
      ag = wmma16(ah1, load_b_frag(frags + FR_WHH, 20 + t, lane), ag);

      v8f ho = load_d(hcur, 64, t * 16, lane);
      v8f hv;
#pragma unroll
      for (int v = 0; v < 8; ++v) {
        float n = tanhf_fast(ai[v] + r[t][v] * ag[v]);
        hv[v] = n + zt[v] * (ho[v] - n);       // (1-z)*n + z*h, 2 ops
      }
      store_d(hnext, 64, t * 16, lane, hv);    // double-buffered: hcur intact
    }

    // ---- dec1: relu(h' @ dec1_w.T + b) -> a1 reuses old-h buffer ----
    v16h an0 = load_a_frag(hnext, 64, 0,  lane);
    v16h an1 = load_a_frag(hnext, 64, 32, lane);
#pragma unroll
    for (int t = 0; t < 4; ++t) {
      v8f acc = bcast8(bd1[t]);
      acc = wmma16(an0, load_b_frag(frags + FR_D1, t,     lane), acc);
      acc = wmma16(an1, load_b_frag(frags + FR_D1, 4 + t, lane), acc);
#pragma unroll
      for (int v = 0; v < 8; ++v) acc[v] = relu1(acc[v]);
      store_d(hcur, 64, t * 16, lane, acc);    // a1 lives in old h buffer
    }

    // ---- dec2: a1 @ dec2_w.T + b  (N=4 valid of 16) ----
    {
      v16h aa0 = load_a_frag(hcur, 64, 0,  lane);
      v16h aa1 = load_a_frag(hcur, 64, 32, lane);
      v8f acc = bcast8(bd2);
      acc = wmma16(aa0, load_b_frag(frags + FR_D2, 0, lane), acc);
      acc = wmma16(aa1, load_b_frag(frags + FR_D2, 1, lane), acc);
      int m0 = (lane & 16) ? 8 : 0;
      if (it == 4) {
        if (nl < 4) {
#pragma unroll
          for (int v = 0; v < 8; ++v) {
            int row = row0 + m0 + v;
            if (row < nrows) out[(size_t)row * 4 + nl] = acc[v];
          }
        }
      } else {
        if (nl < 4) {
#pragma unroll
          for (int v = 0; v < 8; ++v) eny[(m0 + v) * 96 + 64 + nl] = acc[v];
        }
      }
    }

    // swap h buffers (hnext holds h', hcur now holds dead a1)
    float* tmp = hcur; hcur = hnext; hnext = tmp;
  }
}

extern "C" void kernel_launch(void* const* d_in, const int* in_sizes, int n_in,
                              void* d_out, int out_size, void* d_ws, size_t ws_size,
                              hipStream_t stream) {
  const float* x      = (const float*)d_in[0];
  const float* enc_w  = (const float*)d_in[1];
  const float* enc_b  = (const float*)d_in[2];
  const float* ln_g   = (const float*)d_in[3];
  const float* ln_b   = (const float*)d_in[4];
  const float* w_ih   = (const float*)d_in[5];
  const float* w_hh   = (const float*)d_in[6];
  const float* b_ih   = (const float*)d_in[7];
  const float* b_hh   = (const float*)d_in[8];
  const float* dec1_w = (const float*)d_in[9];
  const float* dec1_b = (const float*)d_in[10];
  const float* dec2_w = (const float*)d_in[11];
  const float* dec2_b = (const float*)d_in[12];
  const float* y0     = (const float*)d_in[13];
  float* out = (float*)d_out;

  int nrows = in_sizes[0] / 11;
  int rows_per_block = WAVES * 16;
  int grid = (nrows + rows_per_block - 1) / rows_per_block;

  (void)hipFuncSetAttribute((const void*)feedback_gru_kernel,
                            hipFuncAttributeMaxDynamicSharedMemorySize,
                            (int)SHMEM_BYTES);

  feedback_gru_kernel<<<grid, THREADS, SHMEM_BYTES, stream>>>(
      x, enc_w, enc_b, ln_g, ln_b, w_ih, w_hh, b_ih, b_hh,
      dec1_w, dec1_b, dec2_w, dec2_b, y0, out, nrows);
}
